// SPAIR_22814866276838
// MI455X (gfx1250) — matrix operand order repaired and checked by reading, not compiled, not run on hardware
//
#include <hip/hip_runtime.h>
#include <cstddef>

// ---------------------------------------------------------------------------
// CDNA5 (gfx1250) WMMA implementation of the SPAIR forward pass.
// All GEMMs (4 convs as implicit GEMM + 12 MLP layers per grid cell) run on
// v_wmma_f32_16x16x32_f16. The 8x8 autoregressive cell scan is executed as 22
// anti-diagonal wavefront launches (cells with equal 2h+w are independent).
// Fragments are loaded as 16B-aligned vectors (ds_load_b128/global_load_b128).
// Wave id is readfirstlane'd to an SGPR so per-N-tile guards are scalar
// branches (no exec-mask save/restore around the WMMAs).
// ---------------------------------------------------------------------------

typedef __attribute__((ext_vector_type(16))) _Float16 v16h;
typedef __attribute__((ext_vector_type(8)))  _Float16 v8h;
typedef __attribute__((ext_vector_type(8)))  float    v8f;

#define DEVINL __device__ __forceinline__

DEVINL float sigm_(float x)   { return 1.f / (1.f + expf(-x)); }
DEVINL float softp_(float x)  { return (x > 20.f) ? x : log1pf(expf(x)); }

// --- CDNA5 16-bit WMMA fragment loaders (wave32 layouts per ISA 7.12.2) ----
// A (16x32 f16): lane holds row m=lane&15; elem i -> k = 16*(i>=8) + 8*(lane>=16) + (i&7)
//   => two contiguous 8-half runs at +0 and +16 (both 16B aligned for our strides).
DEVINL v16h fragA16(const _Float16* buf, int stride, int lane) {
    const _Float16* r = buf + (lane & 15) * stride + ((lane & 16) >> 1);
    v8h lo = *(const v8h*)(r);
    v8h hi = *(const v8h*)(r + 16);
    v16h f;
#pragma unroll
    for (int j = 0; j < 8; ++j) { f[j] = lo[j]; f[8 + j] = hi[j]; }
    return f;
}
// B (32x16 f16): lane holds col n=lane&15; elem i -> k = 16*(i>=8) + 8*(lane>=16) + (i&7)
DEVINL v16h fragB16(const _Float16* buf, int stride, int lane) {
    const _Float16* r = buf + (lane & 15) * stride + ((lane & 16) >> 1);
    v8h lo = *(const v8h*)(r);
    v8h hi = *(const v8h*)(r + 16);
    v16h f;
#pragma unroll
    for (int j = 0; j < 8; ++j) { f[j] = lo[j]; f[8 + j] = hi[j]; }
    return f;
}

// LDS-activation x L2-resident-weight GEMM layer. Weights: n-major [n*KPAD+k]
// f16, zero padded; biases f32 zero padded. 4 waves split N tiles; A fragment
// loaded once per K tile and fed to up to 4 register accumulators. `wid` is an
// SGPR (readfirstlane) so tile guards are scalar branches and EXEC stays
// all-ones around every WMMA (ISA 7.12 requirement).
template<int KPAD, int NTILES, int RELU>
DEVINL void gemm_layer(const _Float16* Abuf,
                       const _Float16* W, const float* bias,
                       _Float16* Oh, float* Of, int Ostride,
                       int lane, int wid) {
    constexpr int KT = KPAD / 32;
    v8f acc[4] = {};
    for (int kt = 0; kt < KT; ++kt) {
        v16h a = fragA16(Abuf + kt * 32, KPAD, lane);
#pragma unroll
        for (int q = 0; q < 4; ++q) {
            int nt = wid + q * 4;                 // scalar (SGPR) predicate
            if (nt < NTILES) {
                v16h b = fragB16(W + (size_t)nt * 16 * KPAD + kt * 32, KPAD, lane);
                acc[q] = __builtin_amdgcn_wmma_f32_16x16x32_f16(
                    false, a, false, b, (short)0, acc[q], false, false);
            }
        }
    }
#pragma unroll
    for (int q = 0; q < 4; ++q) {
        int nt = wid + q * 4;
        if (nt < NTILES) {
            int n = nt * 16 + (lane & 15);
            float bv = bias[n];
#pragma unroll
            for (int r = 0; r < 8; ++r) {
                int m = r + ((lane & 16) >> 1);
                float v = acc[q][r] + bv;
                if (RELU) v = fmaxf(v, 0.f);
                if (Oh) Oh[m * Ostride + n] = (_Float16)v;
                if (Of) Of[m * Ostride + n] = v;
            }
        }
    }
}

// ------------------------- weight conversion kernels -----------------------
__global__ void convert_mlp_w(const float* __restrict__ W, const float* __restrict__ b,
                              _Float16* __restrict__ Wo, float* __restrict__ bo,
                              int Kin, int Nout, int Kpad, int Npad) {
    int total = Kpad * Npad;
    for (int idx = blockIdx.x * blockDim.x + threadIdx.x; idx < total + Npad;
         idx += gridDim.x * blockDim.x) {
        if (idx < total) {
            int n = idx / Kpad, k = idx - n * Kpad;
            float v = (n < Nout && k < Kin) ? W[(size_t)k * Nout + n] : 0.f;
            Wo[idx] = (_Float16)v;
        } else {
            int n = idx - total;
            bo[n] = (n < Nout) ? b[n] : 0.f;
        }
    }
}

__global__ void convert_conv_w(const float* __restrict__ W, _Float16* __restrict__ Wo,
                               int Cout, int Cin, int Kpad, int Npad) {
    int total = Kpad * Npad;
    for (int idx = blockIdx.x * blockDim.x + threadIdx.x; idx < total;
         idx += gridDim.x * blockDim.x) {
        int n = idx / Kpad, k = idx - n * Kpad;
        float v = 0.f;
        if (n < Cout && k < 9 * Cin) {
            int kk = k / Cin, ci = k - kk * Cin;
            int kh = kk / 3, kw = kk - kh * 3;
            v = W[(((size_t)n * Cin + ci) * 3 + kh) * 3 + kw];  // OIHW source
        }
        Wo[idx] = (_Float16)v;
    }
}

__global__ void edge_kernel(const float* __restrict__ e, float* __restrict__ out) {
    int i = threadIdx.x;
    if (i < 56) {
        float v = e[i];
        if (i < 4 || i >= 54) v = sigm_(v);
        out[i] = v;
    }
}

// ------------------------- implicit-GEMM conv (WMMA) -----------------------
// One wave computes a 16(M=B*Ho*Wo) x 16(N=Cout) tile; K = 9*Cin (padded).
// Fully specialized: all div/mod become shifts; Cin=64 path loads the
// A-fragment as two contiguous 8-channel global_load_b128's.
template<int CIN, int HIN, int WIN, int HOUT, int WOUT, int COUT, int KPAD,
         bool IN_F32_NCHW, bool OUT_F32>
__global__ __launch_bounds__(32)
void conv_wmma(const void* __restrict__ in,
               const _Float16* __restrict__ wgt, const float* __restrict__ bias,
               void* __restrict__ out) {
    constexpr int HW = HOUT * WOUT;          // power of two for every layer
    constexpr int KT = KPAD / 32;
    constexpr int KR = 9 * CIN;
    const int lane = threadIdx.x;
    const int mbase = blockIdx.x * 16, nbase = blockIdx.y * 16;
    const int m = mbase + (lane & 15);
    const int bidx = m / HW;
    const int rem  = m & (HW - 1);
    const int oh = rem / WOUT, ow = rem & (WOUT - 1);
    const int kb8 = (lane & 16) >> 1;

    v8f acc = {};
    for (int kt = 0; kt < KT; ++kt) {
        v16h a;
        if constexpr (CIN == 64) {
            // KPAD == KR: no zero padding needed; two 8-channel runs per lane.
#pragma unroll
            for (int g = 0; g < 2; ++g) {
                int base = kt * 32 + kb8 + g * 16;
                int kk = base >> 6, ci = base & 63;
                int kh = kk / 3, kw = kk - (kk / 3) * 3;
                int ih = oh * 2 + kh, iw = ow * 2 + kw;    // SAME pad (lo=0,hi=1)
                v8h part = {};
                if (ih < HIN && iw < WIN) {
                    const _Float16* p = (const _Float16*)in +
                        ((((size_t)bidx * HIN + ih) * WIN + iw) << 6) + ci;
                    part = *(const v8h*)p;
                }
#pragma unroll
                for (int j = 0; j < 8; ++j) a[g * 8 + j] = part[j];
            }
        } else {
            // layer 1: fp32 NCHW input, Cin=3, K padded 27->32
#pragma unroll
            for (int i = 0; i < 16; ++i) {
                int k = kt * 32 + (((i & 8) << 1) | kb8 | (i & 7));
                float v = 0.f;
                if (k < KR) {
                    int kk = k / CIN, ci = k - kk * CIN;
                    int kh = kk / 3, kw = kk - (kk / 3) * 3;
                    int ih = oh * 2 + kh, iw = ow * 2 + kw;
                    if (ih < HIN && iw < WIN)
                        v = ((const float*)in)[(((size_t)bidx * CIN + ci) * HIN + ih) * WIN + iw];
                }
                a[i] = (_Float16)v;
            }
        }
        v16h bfr = fragB16(wgt + (size_t)nbase * KPAD + kt * 32, KPAD, lane);
        acc = __builtin_amdgcn_wmma_f32_16x16x32_f16(false, a, false, bfr,
                                                     (short)0, acc, false, false);
    }
    int n = nbase + (lane & 15);
    if (n < COUT) {
        float bv = bias[n];
#pragma unroll
        for (int r = 0; r < 8; ++r) {
            int gm = mbase + r + ((lane & 16) >> 1);
            int b2 = gm / HW;
            int r2 = gm & (HW - 1);
            int oh2 = r2 / WOUT, ow2 = r2 & (WOUT - 1);
            float v = fmaxf(acc[r] + bv, 0.f);
            size_t oidx = (((size_t)b2 * HOUT + oh2) * WOUT + ow2) * COUT + n;
            if (OUT_F32) ((float*)out)[oidx] = v;
            else         ((_Float16*)out)[oidx] = (_Float16)v;
        }
    }
}

// ------------------------------ cell kernel --------------------------------
struct CellArgs {
    const float* x; const float* feat; const float* edge; float* ctx;
    const _Float16* wm[12]; const float* bm[12];
    const float* eps_box; const float* eps_attr; const float* eps_depth; const float* u_pres;
    float* out; int d; int h0;
};

__global__ __launch_bounds__(128)
void cell_kernel(CellArgs A) {
    __shared__ alignas(16) _Float16 sIn[16 * 608];   // widest GEMM input
    __shared__ alignas(16) _Float16 sHid[16 * 256];  // hidden ping-pong partner
    __shared__ float sF1[16 * 112];        // box_out -> z_out -> obj logit
    __shared__ float sF2[16 * 112];        // attr_lat -> attr
    __shared__ float sMisc[16 * 16];       // 0..3 box, 4..7 nbox, 8 depth, 9 pres, 10 pprob
    __shared__ int   sIy0[16 * 14], sIy1[16 * 14], sIx0[16 * 14], sIx1[16 * 14];
    __shared__ float sWy[16 * 14], sWx[16 * 14];

    const int tid = threadIdx.x;
    const int lane = tid & 31;
    // Force the wave id into an SGPR: all per-N-tile guards in gemm_layer
    // become scalar compares + uniform branches (no exec-mask churn).
    const int wid = __builtin_amdgcn_readfirstlane(tid >> 5);
    const int h = A.h0 + (int)blockIdx.y;
    const int w = A.d - 2 * h;
    const int b0 = (int)blockIdx.x * 16;

    // Warm caches for this cell's weights (gfx1250 global_prefetch_b8 path).
#pragma unroll
    for (int i = 0; i < 12; ++i)
        __builtin_prefetch(A.wm[i] + (size_t)tid * 64, 0, 0);

    // cell_feat(100) + context(224) builder (cols 0..323)
    auto common = [&](int s, int c) -> float {
        int b = b0 + s;
        if (c < 100) return A.feat[(((size_t)b * 8 + h) * 8 + w) * 100 + c];
        int cc = c - 100, nb = cc / 56, j = cc - nb * 56;
        int nh = (nb < 3) ? h - 1 : h;
        int nw = (nb == 0) ? w - 1 : (nb == 1) ? w : (nb == 2) ? w + 1 : w - 1;
        if (nh < 0 || nw < 0 || nw > 7) return A.edge[j];
        return A.ctx[(((size_t)b * 8 + nh) * 8 + nw) * 56 + j];
    };

    // ---- box MLP: [feat,ctx](324->352) -> 100(128) -> 100(128) -> 108(112)
    for (int idx = tid; idx < 16 * 352; idx += 128) {
        int s = idx / 352, c = idx - s * 352;
        sIn[idx] = (_Float16)((c < 324) ? common(s, c) : 0.f);
    }
    __syncthreads();
    gemm_layer<352, 8, 1>(sIn, A.wm[0], A.bm[0], sHid, nullptr, 128, lane, wid);
    __syncthreads();
    gemm_layer<128, 8, 1>(sHid, A.wm[1], A.bm[1], sIn, nullptr, 128, lane, wid);
    __syncthreads();
    gemm_layer<128, 7, 0>(sIn, A.wm[2], A.bm[2], nullptr, sF1, 112, lane, wid);
    __syncthreads();

    // ---- box sampling (16 samples x 4 dims)
    if (tid < 64) {
        int s = tid >> 2, dc = tid & 3, b = b0 + s;
        float mv = sF1[s * 112 + dc], sr = sF1[s * 112 + 4 + dc];
        float sp = softp_(sr) + 1e-4f;
        float eps = A.eps_box[(((size_t)b * 4 + dc) * 8 + h) * 8 + w];
        float sg = sigm_(mv + sp * eps);
        sMisc[s * 16 + dc] = sg;
        float nb = (dc == 0) ? (h + sg) * (16.f / 128.f)
                 : (dc == 1) ? (w + sg) * (16.f / 128.f)
                 : sg * (48.f / 128.f);
        sMisc[s * 16 + 4 + dc] = nb;
    }
    __syncthreads();

    // ---- glimpse bilinear coords (16 x 14 x {y,x})
    for (int t = tid; t < 448; t += 128) {
        int s = t / 28, r = t - s * 28;
        int axis = r / 14, j = r - axis * 14;
        float lin = (float)j * (1.f / 13.f) - 0.5f;
        float p = (sMisc[s * 16 + 4 + axis] + lin * sMisc[s * 16 + 6 + axis]) * 127.f;
        p = fminf(fmaxf(p, 0.f), 127.f);
        float p0 = floorf(p);
        int i0 = (int)p0, i1 = i0 + 1; if (i1 > 127) i1 = 127;
        float wt = p - p0;
        if (axis == 0) { sIy0[s*14+j]=i0; sIy1[s*14+j]=i1; sWy[s*14+j]=wt; }
        else           { sIx0[s*14+j]=i0; sIx1[s*14+j]=i1; sWx[s*14+j]=wt; }
    }
    __syncthreads();

    // ---- glimpse gather into sIn (588 -> 608)
    for (int idx = tid; idx < 16 * 608; idx += 128) {
        int s = idx / 608, c = idx - s * 608;
        float v = 0.f;
        if (c < 588) {
            int ch = c % 3, t2 = c / 3, gi = t2 % 14, gj = t2 / 14;
            int b = b0 + s;
            int y0 = sIy0[s*14+gj], y1 = sIy1[s*14+gj];
            int x0 = sIx0[s*14+gi], x1 = sIx1[s*14+gi];
            float wy = sWy[s*14+gj], wx = sWx[s*14+gi];
            const float* img = A.x + ((size_t)b * 3 + ch) * 128 * 128;  // NCHW
            float g00 = img[y0*128+x0], g01 = img[y0*128+x1];
            float g10 = img[y1*128+x0], g11 = img[y1*128+x1];
            v = g00*(1-wy)*(1-wx) + g01*(1-wy)*wx + g10*wy*(1-wx) + g11*wy*wx;
        }
        sIn[idx] = (_Float16)v;
    }
    __syncthreads();

    // ---- enc MLP: 588(608) -> 256 -> 128 -> 100(112)
    gemm_layer<608, 16, 1>(sIn, A.wm[3], A.bm[3], sHid, nullptr, 256, lane, wid);
    __syncthreads();
    gemm_layer<256, 8, 1>(sHid, A.wm[4], A.bm[4], sIn, nullptr, 128, lane, wid);
    __syncthreads();
    gemm_layer<128, 7, 0>(sIn, A.wm[5], A.bm[5], nullptr, sF2, 112, lane, wid);
    __syncthreads();

    // ---- attr sampling (writes cols 0..49 of sF2, reads 50..99)
    for (int t = tid; t < 800; t += 128) {
        int s = t / 50, a = t - s * 50, b = b0 + s;
        float sp = softp_(sF2[s * 112 + 50 + a]) + 1e-4f;
        float eps = A.eps_attr[(((size_t)b * 50 + a) * 8 + h) * 8 + w];
        float attr = sF2[s * 112 + a] + sp * eps;
        sF2[s * 112 + a] = attr;
        A.out[(((size_t)b * 57 + 4 + a) * 8 + h) * 8 + w] = attr;
    }
    __syncthreads();

    // ---- z MLP: [feat,ctx,passthru,box,attr](478->480) -> 100 -> 100 -> 102(112)
    for (int idx = tid; idx < 16 * 480; idx += 128) {
        int s = idx / 480, c = idx - s * 480;
        float v;
        if      (c < 324) v = common(s, c);
        else if (c < 424) v = sF1[s * 112 + 8 + (c - 324)];   // passthru
        else if (c < 428) v = sMisc[s * 16 + (c - 424)];       // box
        else if (c < 478) v = sF2[s * 112 + (c - 428)];        // attr
        else              v = 0.f;
        sIn[idx] = (_Float16)v;
    }
    __syncthreads();
    gemm_layer<480, 8, 1>(sIn, A.wm[6], A.bm[6], sHid, nullptr, 128, lane, wid);
    __syncthreads();
    gemm_layer<128, 8, 1>(sHid, A.wm[7], A.bm[7], sIn, nullptr, 128, lane, wid);
    __syncthreads();
    gemm_layer<128, 7, 0>(sIn, A.wm[8], A.bm[8], nullptr, sF1, 112, lane, wid);
    __syncthreads();

    // ---- depth sampling
    if (tid < 16) {
        int s = tid, b = b0 + s;
        float sp = softp_(sF1[s * 112 + 1]) + 1e-4f;
        float dl = sF1[s * 112 + 0] + sp * A.eps_depth[(size_t)b * 64 + h * 8 + w];
        dl = fminf(fmaxf(dl, -10.f), 10.f);
        sMisc[s * 16 + 8] = 4.f * sigm_(dl);
    }
    __syncthreads();

    // ---- obj MLP: [feat,ctx,passthru2,box,attr,depth](479->480) -> 100 -> 100 -> 1(16)
    for (int idx = tid; idx < 16 * 480; idx += 128) {
        int s = idx / 480, c = idx - s * 480;
        float v;
        if      (c < 324) v = common(s, c);
        else if (c < 424) v = sF1[s * 112 + 2 + (c - 324)];   // passthru2
        else if (c < 428) v = sMisc[s * 16 + (c - 424)];       // box
        else if (c < 478) v = sF2[s * 112 + (c - 428)];        // attr
        else if (c == 478) v = sMisc[s * 16 + 8];              // depth
        else              v = 0.f;
        sIn[idx] = (_Float16)v;
    }
    __syncthreads();
    gemm_layer<480, 8, 1>(sIn, A.wm[9],  A.bm[9],  sHid, nullptr, 128, lane, wid);
    __syncthreads();
    gemm_layer<128, 8, 1>(sHid, A.wm[10], A.bm[10], sIn, nullptr, 128, lane, wid);
    __syncthreads();
    gemm_layer<128, 1, 0>(sIn, A.wm[11], A.bm[11], nullptr, sF1, 112, lane, wid);
    __syncthreads();

    // ---- presence
    if (tid < 16) {
        int s = tid, b = b0 + s;
        float plog = fminf(fmaxf(sF1[s * 112 + 0], -10.f), 10.f);
        float u = A.u_pres[(size_t)b * 64 + h * 8 + w];
        sMisc[s * 16 + 9]  = sigm_(plog + logf(u) - log1pf(-u));  // pres
        sMisc[s * 16 + 10] = sigm_(plog);                          // pprob
    }
    __syncthreads();

    // ---- outputs + ctx for this cell
    for (int t = tid; t < 16 * 7; t += 128) {
        int s = t / 7, q = t - s * 7, b = b0 + s;
        float v; int ch;
        if (q < 4)      { v = sMisc[s * 16 + 4 + q]; ch = q;  }   // nbox
        else if (q == 4){ v = sMisc[s * 16 + 8];     ch = 54; }   // depth
        else if (q == 5){ v = sMisc[s * 16 + 9];     ch = 55; }   // pres
        else            { v = sMisc[s * 16 + 10];    ch = 56; }   // pprob
        A.out[(((size_t)b * 57 + ch) * 8 + h) * 8 + w] = v;
    }
    for (int t = tid; t < 16 * 56; t += 128) {
        int s = t / 56, j = t - s * 56, b = b0 + s;
        float v;
        if      (j < 4)  v = sMisc[s * 16 + j];        // box (sigmoids)
        else if (j < 54) v = sF2[s * 112 + (j - 4)];   // attr
        else if (j == 54) v = sMisc[s * 16 + 8];       // depth
        else              v = sMisc[s * 16 + 9];       // pres
        A.ctx[(((size_t)b * 8 + h) * 8 + w) * 56 + j] = v;
    }
}

// ------------------------------- host side ---------------------------------
extern "C" void kernel_launch(void* const* d_in, const int* in_sizes, int n_in,
                              void* d_out, int out_size, void* d_ws, size_t ws_size,
                              hipStream_t stream) {
    (void)n_in; (void)out_size; (void)ws_size;
    auto g = [&](int i) { return (const float*)d_in[i]; };
    const float* X = g(0);
    const float *BW[3], *BB[3], *CW[4], *CB[4], *EW[3], *EB[3], *ZW[3], *ZB[3],
                *OW[3], *OB[3], *EDGE;
    // Param flattening order detection: sorted dict keys (box first, W0=324*100)
    // vs insertion order (conv first, W0=64*3*3*3).
    if (in_sizes[1] == 324 * 100) {
        for (int i = 0; i < 3; i++) { BW[i]=g(1+2*i);  BB[i]=g(2+2*i);  }
        for (int i = 0; i < 4; i++) { CW[i]=g(7+2*i);  CB[i]=g(8+2*i);  }
        EDGE = g(15);
        for (int i = 0; i < 3; i++) { EW[i]=g(16+2*i); EB[i]=g(17+2*i); }
        for (int i = 0; i < 3; i++) { OW[i]=g(22+2*i); OB[i]=g(23+2*i); }
        for (int i = 0; i < 3; i++) { ZW[i]=g(28+2*i); ZB[i]=g(29+2*i); }
    } else {
        for (int i = 0; i < 4; i++) { CW[i]=g(1+2*i);  CB[i]=g(2+2*i);  }
        for (int i = 0; i < 3; i++) { BW[i]=g(9+2*i);  BB[i]=g(10+2*i); }
        for (int i = 0; i < 3; i++) { EW[i]=g(15+2*i); EB[i]=g(16+2*i); }
        for (int i = 0; i < 3; i++) { ZW[i]=g(21+2*i); ZB[i]=g(22+2*i); }
        for (int i = 0; i < 3; i++) { OW[i]=g(27+2*i); OB[i]=g(28+2*i); }
        EDGE = g(33);
    }
    const float* EPSB = g(34); const float* EPSA = g(35);
    const float* EPSD = g(36); const float* UPR  = g(37);

    // -------- workspace carving --------
    size_t off = 0;
    auto carve = [&](size_t bytes) -> char* {
        char* p = (char*)d_ws + off;
        off += (bytes + 255) & ~(size_t)255;
        return p;
    };
    _Float16* wc[4];
    wc[0] = (_Float16*)carve((size_t)32  * 64  * 2);
    wc[1] = (_Float16*)carve((size_t)576 * 64  * 2);
    wc[2] = (_Float16*)carve((size_t)576 * 64  * 2);
    wc[3] = (_Float16*)carve((size_t)576 * 112 * 2);
    struct LW { int Kin, Nout, Kpad, Npad; };
    const LW lw[12] = {
        {324,100,352,128},{100,100,128,128},{100,108,128,112},   // box
        {588,256,608,256},{256,128,256,128},{128,100,128,112},   // enc
        {478,100,480,128},{100,100,128,128},{100,102,128,112},   // z
        {479,100,480,128},{100,100,128,128},{100,  1,128, 16}};  // obj
    const float* srcW[12] = {BW[0],BW[1],BW[2], EW[0],EW[1],EW[2],
                             ZW[0],ZW[1],ZW[2], OW[0],OW[1],OW[2]};
    const float* srcB[12] = {BB[0],BB[1],BB[2], EB[0],EB[1],EB[2],
                             ZB[0],ZB[1],ZB[2], OB[0],OB[1],OB[2]};
    _Float16* wm[12]; float* bm[12];
    for (int i = 0; i < 12; i++) {
        wm[i] = (_Float16*)carve((size_t)lw[i].Kpad * lw[i].Npad * 2);
        bm[i] = (float*)carve((size_t)lw[i].Npad * 4);
    }
    float* edgev = (float*)carve(56 * 4);
    _Float16* act1 = (_Float16*)carve((size_t)256 * 64 * 64 * 64 * 2);
    _Float16* act2 = (_Float16*)carve((size_t)256 * 32 * 32 * 64 * 2);
    _Float16* act3 = (_Float16*)carve((size_t)256 * 16 * 16 * 64 * 2);
    float* featb = (float*)carve((size_t)256 * 64 * 100 * 4);
    float* ctxb  = (float*)carve((size_t)256 * 64 * 56 * 4);

    // -------- weight conversion (fp32 -> fp16 fragment-native layout) --------
    for (int i = 0; i < 12; i++) {
        int total = lw[i].Kpad * lw[i].Npad + lw[i].Npad;
        convert_mlp_w<<<(total + 255) / 256, 256, 0, stream>>>(
            srcW[i], srcB[i], wm[i], bm[i], lw[i].Kin, lw[i].Nout, lw[i].Kpad, lw[i].Npad);
    }
    const int cCout[4] = {64, 64, 64, 100}, cCin[4] = {3, 64, 64, 64};
    const int cKpad[4] = {32, 576, 576, 576}, cNpad[4] = {64, 64, 64, 112};
    for (int i = 0; i < 4; i++) {
        int total = cKpad[i] * cNpad[i];
        convert_conv_w<<<(total + 255) / 256, 256, 0, stream>>>(
            CW[i], wc[i], cCout[i], cCin[i], cKpad[i], cNpad[i]);
    }
    edge_kernel<<<1, 64, 0, stream>>>(EDGE, edgev);

    // -------- backbone: 4 implicit-GEMM WMMA convs (NHWC fp16 activations) ---
    conv_wmma<3, 128, 128, 64, 64, 64, 32, true, false>
        <<<dim3(256*64*64/16, 4), 32, 0, stream>>>(X, wc[0], CB[0], act1);
    conv_wmma<64, 64, 64, 32, 32, 64, 576, false, false>
        <<<dim3(256*32*32/16, 4), 32, 0, stream>>>(act1, wc[1], CB[1], act2);
    conv_wmma<64, 32, 32, 16, 16, 64, 576, false, false>
        <<<dim3(256*16*16/16, 4), 32, 0, stream>>>(act2, wc[2], CB[2], act3);
    conv_wmma<64, 16, 16, 8, 8, 100, 576, false, true>
        <<<dim3(256*8*8/16, 7), 32, 0, stream>>>(act3, wc[3], CB[3], featb);

    // -------- autoregressive cell scan: 22 anti-diagonal wavefront steps -----
    for (int dd = 0; dd < 22; ++dd) {
        int hlo = (dd <= 7) ? 0 : (dd - 6) / 2;
        int hhi = (dd / 2 < 7) ? dd / 2 : 7;
        int cnt = hhi - hlo + 1;
        CellArgs a;
        a.x = X; a.feat = featb; a.edge = edgev; a.ctx = ctxb;
        for (int i = 0; i < 12; i++) { a.wm[i] = wm[i]; a.bm[i] = bm[i]; }
        a.eps_box = EPSB; a.eps_attr = EPSA; a.eps_depth = EPSD; a.u_pres = UPR;
        a.out = (float*)d_out; a.d = dd; a.h0 = hlo;
        cell_kernel<<<dim3(16, cnt), 128, 0, stream>>>(a);
    }
}
